// GraphSAGE_64957085385410
// MI455X (gfx1250) — compile-verified, matrix operand order checked by reading, not run on hardware
//
#include <hip/hip_runtime.h>
#include <hip/hip_bf16.h>

// ---------- types for WMMA ----------
typedef __bf16 v16bf __attribute__((ext_vector_type(16)));
typedef __bf16 v8bf  __attribute__((ext_vector_type(8)));
typedef float  v8f   __attribute__((ext_vector_type(8)));

#define GN 100000   // nodes (100000 % 16 == 0 -> 6250 M-tiles)
#define GE 500000   // edges
#define GH 32       // hidden

// Load a 16-element bf16 B-fragment (two aligned 16B chunks) from a [n][Kpad] row.
static __device__ __forceinline__ v16bf ld_bfrag(const __bf16* __restrict__ row,
                                                 int k1, int k2) {
    v8bf lo = *reinterpret_cast<const v8bf*>(row + k1);
    v8bf hi = *reinterpret_cast<const v8bf*>(row + k2);
    return __builtin_shufflevector(lo, hi,
        0, 1, 2, 3, 4, 5, 6, 7, 8, 9, 10, 11, 12, 13, 14, 15);
}

// ---------------------------------------------------------------------------
// Pre-convert weights: Wb[64][Kpad] (bf16, zero-padded K tail).
// Rows 0..31 = Wl, rows 32..63 = Wr. Run once per layer; tiny.
// ---------------------------------------------------------------------------
__global__ void convert_w(const float* __restrict__ Wl,
                          const float* __restrict__ Wr,
                          __bf16* __restrict__ Wb,
                          int K, int Kpad)
{
    int idx = blockIdx.x * blockDim.x + threadIdx.x;
    if (idx >= 64 * Kpad) return;
    int n = idx / Kpad;
    int k = idx - n * Kpad;
    const float* row = (n < 32) ? (Wl + (size_t)n * K) : (Wr + (size_t)(n - 32) * K);
    Wb[idx] = (k < K) ? (__bf16)row[k] : (__bf16)0.0f;
}

// ---------------------------------------------------------------------------
// Fused projection GEMM:  Y[M,64] = Xin[M,K] @ Wb.T   (Wb is bf16 [64][Kpad])
// One wave per 16-row stripe computes ALL FOUR 16x16 N-tiles so the X stream
// is read from HBM exactly once (X >> L2, W << L2). 4x v_wmma per K-step.
// grid = (M/16), block = 32 (1 wave).
// ---------------------------------------------------------------------------
__global__ void __launch_bounds__(32)
sage_gemm_wmma(const float* __restrict__ Xin,
               const __bf16* __restrict__ Wb,
               float* __restrict__ Y,
               int K, int Kpad)
{
    const int lane = threadIdx.x & 31;
    const int half = lane >> 4;       // 0: lanes 0-15, 1: lanes 16-31
    const int r    = lane & 15;

    const int m0 = blockIdx.x << 4;   // tile rows m0..m0+15

    const float* __restrict__ arow = Xin + (size_t)(m0 + r) * K;
    const __bf16* __restrict__ b0 = Wb + (size_t)(r)      * Kpad;
    const __bf16* __restrict__ b1 = Wb + (size_t)(16 + r) * Kpad;
    const __bf16* __restrict__ b2 = Wb + (size_t)(32 + r) * Kpad;
    const __bf16* __restrict__ b3 = Wb + (size_t)(48 + r) * Kpad;

    v8f c0 = {}, c1 = {}, c2 = {}, c3 = {};

    for (int kb = 0; kb < K; kb += 32) {
        // ISA A-fragment layout (16-bit 16x32): lane half selects K sub-block.
        const int k1 = kb + half * 8;        // elements 0..7  -> K = k1..k1+7
        const int k2 = kb + 16 + half * 8;   // elements 8..15 -> K = k2..k2+7

        v16bf a;
        if (kb + 32 <= K) {
            #pragma unroll
            for (int i = 0; i < 8; ++i) {
                // X is a pure 573MB stream: non-temporal so it can't evict
                // the hot W rows / the Y tile we re-gather next kernel.
                a[i]     = (__bf16)__builtin_nontemporal_load(arow + k1 + i);
                a[i + 8] = (__bf16)__builtin_nontemporal_load(arow + k2 + i);
            }
        } else {  // K tail (1433 % 32 == 25): zero-pad
            #pragma unroll
            for (int i = 0; i < 8; ++i) {
                int ka = k1 + i, kc = k2 + i;
                a[i]     = (__bf16)(ka < K ? arow[ka] : 0.0f);
                a[i + 8] = (__bf16)(kc < K ? arow[kc] : 0.0f);
            }
        }

        // Weight fragments: pre-converted bf16, Kpad%32==0 -> no guards,
        // 16B-aligned vector loads.
        v16bf B0 = ld_bfrag(b0, k1, k2);
        v16bf B1 = ld_bfrag(b1, k1, k2);
        v16bf B2 = ld_bfrag(b2, k1, k2);
        v16bf B3 = ld_bfrag(b3, k1, k2);

        c0 = __builtin_amdgcn_wmma_f32_16x16x32_bf16(false, a, false, B0, (short)0, c0, false, false);
        c1 = __builtin_amdgcn_wmma_f32_16x16x32_bf16(false, a, false, B1, (short)0, c1, false, false);
        c2 = __builtin_amdgcn_wmma_f32_16x16x32_bf16(false, a, false, B2, (short)0, c2, false, false);
        c3 = __builtin_amdgcn_wmma_f32_16x16x32_bf16(false, a, false, B3, (short)0, c3, false, false);
    }

    // C/D f32 layout: VGPR j -> row (j + 8*half), column = lane&15
    #pragma unroll
    for (int j = 0; j < 8; ++j) {
        size_t base = (size_t)(m0 + j + (half << 3)) * 64 + r;
        Y[base]      = c0[j];
        Y[base + 16] = c1[j];
        Y[base + 32] = c2[j];
        Y[base + 48] = c3[j];
    }
}

// ---------------------------------------------------------------------------
__global__ void zero_f32(float* __restrict__ p, long long n)
{
    long long i = (long long)blockIdx.x * blockDim.x + threadIdx.x;
    if (i < n) p[i] = 0.0f;
}

// One lane per (edge, feature). Scatter-add projected messages; lane f==0
// optionally accumulates in-degree counts. Y rows live in L2 (just written).
__global__ void edge_aggregate(const float* __restrict__ Y,   // [N,64], cols 0..31
                               const int* __restrict__ src,
                               const int* __restrict__ tgt,
                               float* __restrict__ agg,       // [N,32]
                               float* __restrict__ cnt,       // [N] or nullptr
                               int E)
{
    long long idx = (long long)blockIdx.x * blockDim.x + threadIdx.x;
    if (idx >= (long long)E * GH) return;
    int e = (int)(idx >> 5);
    int f = (int)(idx & 31);
    int s = src[e];
    int t = tgt[e];
    float v = Y[(size_t)s * 64 + f];
    unsafeAtomicAdd(&agg[(size_t)t * GH + f], v);
    if (cnt && f == 0) unsafeAtomicAdd(&cnt[t], 1.0f);
}

// h1 = agg/max(cnt,1) + b + Y[:,32:64]
__global__ void finalize_layer(const float* __restrict__ agg,
                               const float* __restrict__ cnt,
                               const float* __restrict__ bias,
                               const float* __restrict__ Y,
                               float* __restrict__ h,
                               int N)
{
    long long idx = (long long)blockIdx.x * blockDim.x + threadIdx.x;
    if (idx >= (long long)N * GH) return;
    int nd = (int)(idx >> 5);
    int f  = (int)(idx & 31);
    float d = fmaxf(cnt[nd], 1.0f);
    h[idx] = agg[idx] / d + bias[f] + Y[(size_t)nd * 64 + 32 + f];
}

// Final layer: mean-combine + bias + right term, relu, log_softmax over 32
// features. One wave32 per node: lane == feature, shfl_xor reductions.
__global__ void finalize_softmax(const float* __restrict__ agg,
                                 const float* __restrict__ cnt,
                                 const float* __restrict__ bias,
                                 const float* __restrict__ Y,
                                 float* __restrict__ out,
                                 int N)
{
    int nd = blockIdx.x * (blockDim.x >> 5) + (threadIdx.x >> 5);
    int f  = threadIdx.x & 31;
    if (nd >= N) return;
    float d = fmaxf(cnt[nd], 1.0f);
    float v = agg[(size_t)nd * GH + f] / d + bias[f] + Y[(size_t)nd * 64 + 32 + f];
    v = fmaxf(v, 0.0f);   // relu

    float m = v;
    #pragma unroll
    for (int off = 16; off > 0; off >>= 1)
        m = fmaxf(m, __shfl_xor(m, off, 32));
    float ex = __expf(v - m);
    float s = ex;
    #pragma unroll
    for (int off = 16; off > 0; off >>= 1)
        s += __shfl_xor(s, off, 32);
    out[(size_t)nd * GH + f] = (v - m) - __logf(s);
}

// ---------------------------------------------------------------------------
extern "C" void kernel_launch(void* const* d_in, const int* in_sizes, int n_in,
                              void* d_out, int out_size, void* d_ws, size_t ws_size,
                              hipStream_t stream)
{
    const float* x    = (const float*)d_in[0];   // [N,1433]
    const int*   eidx = (const int*)  d_in[1];   // [2,E]
    const float* W1l  = (const float*)d_in[2];   // [32,1433]
    const float* b1l  = (const float*)d_in[3];   // [32]
    const float* W1r  = (const float*)d_in[4];   // [32,1433]
    const float* W2l  = (const float*)d_in[5];   // [32,32]
    const float* b2l  = (const float*)d_in[6];   // [32]
    const float* W2r  = (const float*)d_in[7];   // [32,32]
    float* out = (float*)d_out;

    const int N = GN, E = GE;
    const int K1 = 1433, K1p = 1440;   // pad to multiple of 32
    const int K2 = GH,   K2p = GH;
    const int* src = eidx;
    const int* tgt = eidx + E;

    // workspace layout (floats; ws base is 256B aligned, all offsets 16B aligned)
    float* ws   = (float*)d_ws;
    float* Y1   = ws;                          // N*64
    float* Y2   = Y1 + (size_t)N * 64;         // N*64
    float* h1   = Y2 + (size_t)N * 64;         // N*32
    float* agg  = h1 + (size_t)N * 32;         // N*32
    float* cnt  = agg + (size_t)N * 32;        // N (contiguous after agg)
    __bf16* Wb1 = (__bf16*)(ws + (size_t)N * 192 + (size_t)N); // 64*K1p bf16
    __bf16* Wb2 = Wb1 + (size_t)64 * K1p;                      // 64*K2p bf16

    const int tpb = 256;

    // 0) weight pre-conversion to padded bf16 (tiny; done inside capture,
    //    deterministic every call)
    convert_w<<<(64 * K1p + tpb - 1) / tpb, tpb, 0, stream>>>(W1l, W1r, Wb1, K1, K1p);
    convert_w<<<(64 * K2p + tpb - 1) / tpb, tpb, 0, stream>>>(W2l, W2r, Wb2, K2, K2p);

    // 1) zero agg + cnt (contiguous N*33 floats)
    {
        long long n = (long long)N * 33;
        zero_f32<<<(unsigned)((n + tpb - 1) / tpb), tpb, 0, stream>>>(agg, n);
    }

    // 2) layer-1 fused projection: Y1 = x @ [W1l;W1r].T  (project BEFORE aggregate)
    sage_gemm_wmma<<<N / 16, 32, 0, stream>>>(x, Wb1, Y1, K1, K1p);

    // 3) aggregate projected messages (32-dim) + degree counts
    {
        long long n = (long long)E * GH;
        edge_aggregate<<<(unsigned)((n + tpb - 1) / tpb), tpb, 0, stream>>>(
            Y1, src, tgt, agg, cnt, E);
    }

    // 4) h1 = agg/deg + b1l + Y1[:,32:]
    {
        long long n = (long long)N * GH;
        finalize_layer<<<(unsigned)((n + tpb - 1) / tpb), tpb, 0, stream>>>(
            agg, cnt, b1l, Y1, h1, N);
    }

    // 5) re-zero agg for layer 2
    {
        long long n = (long long)N * GH;
        zero_f32<<<(unsigned)((n + tpb - 1) / tpb), tpb, 0, stream>>>(agg, n);
    }

    // 6) layer-2 fused projection: Y2 = h1 @ [W2l;W2r].T  (K=32, one WMMA step)
    sage_gemm_wmma<<<N / 16, 32, 0, stream>>>(h1, Wb2, Y2, K2, K2p);

    // 7) aggregate layer-2 messages
    {
        long long n = (long long)E * GH;
        edge_aggregate<<<(unsigned)((n + tpb - 1) / tpb), tpb, 0, stream>>>(
            Y2, src, tgt, agg, nullptr, E);
    }

    // 8) combine + relu + log_softmax -> out
    {
        int nodes_per_block = tpb / 32;
        int blocks = (N + nodes_per_block - 1) / nodes_per_block;
        finalize_softmax<<<blocks, tpb, 0, stream>>>(agg, cnt, b2l, Y2, out, N);
    }
}
